// ArcFace_20624432955551
// MI455X (gfx1250) — matrix operand order repaired and checked by reading, not compiled, not run on hardware
//
#include <hip/hip_runtime.h>
#include <hip/hip_bf16.h>
#include <math.h>

typedef __attribute__((ext_vector_type(16))) _Float16 v16h;
typedef __attribute__((ext_vector_type(8)))  _Float16 h8;
typedef __attribute__((ext_vector_type(8)))  float    v8f;

#define DDIM 512
#define CCLS 1024

// ---------------- kernel 0: zero counters / accumulators ----------------
__global__ void k_init(int* __restrict__ counts, float* __restrict__ acc) {
    int t = threadIdx.x;
    counts[t] = 0;
    acc[t]    = 0.0f;
}

// ---------------- kernel 1: per-row inv-norm + class histogram ----------
// one wave (32 lanes) per row; 512 floats = 4 x float4 per lane, coalesced
__global__ void k_norm_count(const float* __restrict__ X, const int* __restrict__ lbl,
                             float* __restrict__ rnorm, int* __restrict__ counts) {
    int lane = threadIdx.x & 31;
    int wave = threadIdx.x >> 5;
    int row  = blockIdx.x * 8 + wave;
    const float4* Xr = (const float4*)(X + (size_t)row * DDIM);
    float s = 0.0f;
#pragma unroll
    for (int k = 0; k < 4; ++k) {
        float4 v = Xr[lane + 32 * k];
        s += v.x * v.x + v.y * v.y + v.z * v.z + v.w * v.w;
    }
#pragma unroll
    for (int o = 16; o > 0; o >>= 1) s += __shfl_xor(s, o);
    if (lane == 0) {
        rnorm[row] = 1.0f / sqrtf(s);
        atomicAdd(&counts[lbl[row]], 1);
    }
}

// ---------------- kernel 2: exclusive prefix scan of counts -------------
__global__ void k_scan(const int* __restrict__ counts, int* __restrict__ offsets,
                       int* __restrict__ fill) {
    __shared__ int sh[CCLS];
    int t  = threadIdx.x;
    int c0 = counts[t];
    sh[t]  = c0;
    __syncthreads();
    for (int off = 1; off < CCLS; off <<= 1) {
        int v = (t >= off) ? sh[t - off] : 0;
        __syncthreads();
        sh[t] += v;
        __syncthreads();
    }
    int excl   = sh[t] - c0;   // exclusive scan
    offsets[t] = excl;
    fill[t]    = excl;
}

// ---------------- kernel 3: counting-sort scatter of row ids ------------
__global__ void k_scatter(const int* __restrict__ lbl, int* __restrict__ fill,
                          int* __restrict__ rowidx, int N) {
    int i = blockIdx.x * blockDim.x + threadIdx.x;
    if (i >= N) return;
    int pos = atomicAdd(&fill[lbl[i]], 1);
    rowidx[pos] = i;
}

// ---------------- kernel 4: class prototypes W (f16) + S_c --------------
// one workgroup per class; thread t owns columns t and t+256
__global__ void k_protos(const float* __restrict__ X, const float* __restrict__ rnorm,
                         const int* __restrict__ rowidx, const int* __restrict__ offsets,
                         const int* __restrict__ counts,
                         _Float16* __restrict__ Wh, float* __restrict__ Ssum) {
    __shared__ float red[256];
    int c = blockIdx.x, t = threadIdx.x;
    int cnt = counts[c], start = offsets[c];
    float a0 = 0.0f, a1 = 0.0f;
    for (int j = 0; j < cnt; ++j) {
        int row = rowidx[start + j];
        float rn = rnorm[row];
        const float* xr = X + (size_t)row * DDIM;
        a0 = fmaf(xr[t], rn, a0);
        a1 = fmaf(xr[t + 256], rn, a1);
    }
    float ic = 1.0f / (float)max(cnt, 1);
    float v0 = a0 * ic, v1 = a1 * ic;
    // ||V||^2 block reduce
    red[t] = v0 * v0 + v1 * v1;
    __syncthreads();
    for (int o = 128; o > 0; o >>= 1) { if (t < o) red[t] += red[t + o]; __syncthreads(); }
    float wn  = sqrtf(red[0]);
    __syncthreads();
    float inv = (wn > 0.0f) ? (1.0f / wn) : 1.0f;
    float w0 = v0 * inv, w1 = v1 * inv;
    // S_c = sum_d W
    red[t] = w0 + w1;
    __syncthreads();
    for (int o = 128; o > 0; o >>= 1) { if (t < o) red[t] += red[t + o]; __syncthreads(); }
    if (t == 0) Ssum[c] = red[0];
    Wh[(size_t)c * DDIM + t]       = (_Float16)w0;
    Wh[(size_t)c * DDIM + t + 256] = (_Float16)w1;
}

// ---------------- kernel 5: logZ + gathered dot via WMMA ----------------
// one wave per 16-row tile. Lanes l and l+16 jointly hold row (l&15):
//   A (16x32 f16):  lane<16 -> K {0..7,16..23}+32t ; lane>=16 -> K {8..15,24..31}+32t
//   B (32x16 f16):  col N = lane&15 = W[label]; lane<16 -> K {0..7,16..23}+32t ; else K {8..15,24..31}+32t
// Diagonal of D = dot(Xn_i, W[label_i]); owners: lanes 0..7 (rows 0..7, elem=lane),
// lanes 24..31 (rows 8..15, elem=lane-24).
__global__ void k_ce(const float* __restrict__ X, const int* __restrict__ lbl,
                     const float* __restrict__ rnorm, const _Float16* __restrict__ Wh,
                     const float* __restrict__ Ssum, float* __restrict__ acc) {
    int lane = threadIdx.x & 31;
    int wave = threadIdx.x >> 5;
    int tile = blockIdx.x * 8 + wave;
    int base = tile * 16;
    int r    = base + (lane & 15);
    int cls  = lbl[r];
    float rn = rnorm[r];
    const float*    xr = X  + (size_t)r   * DDIM;
    const _Float16* wr = Wh + (size_t)cls * DDIM;
    bool hi = lane >= 16;

    float m = -3.0e38f, s = 0.0f;   // online softmax state (f32)
    v8f cacc = {};
    for (int t = 0; t < 16; ++t) {
        int k0 = 32 * t + (hi ? 8 : 0);
        int k1 = k0 + 16;
        float4 x0 = *(const float4*)(xr + k0);
        float4 x1 = *(const float4*)(xr + k0 + 4);
        float4 x2 = *(const float4*)(xr + k1);
        float4 x3 = *(const float4*)(xr + k1 + 4);
        float f[16] = { x0.x, x0.y, x0.z, x0.w, x1.x, x1.y, x1.z, x1.w,
                        x2.x, x2.y, x2.z, x2.w, x3.x, x3.y, x3.z, x3.w };
        float cm = f[0] * rn;
#pragma unroll
        for (int j = 0; j < 16; ++j) { f[j] *= rn; cm = fmaxf(cm, f[j]); }
        float M2 = fmaxf(m, cm);
        s *= __expf(m - M2);
#pragma unroll
        for (int j = 0; j < 16; ++j) s += __expf(f[j] - M2);
        m = M2;

        v16h a;
#pragma unroll
        for (int j = 0; j < 16; ++j) a[j] = (_Float16)f[j];

        int kb = 32 * t + (hi ? 8 : 0);
        h8 w0 = *(const h8*)(wr + kb);
        h8 w1 = *(const h8*)(wr + kb + 16);
        v16h b;
#pragma unroll
        for (int j = 0; j < 8; ++j) { b[j] = w0[j]; b[j + 8] = w1[j]; }

        cacc = __builtin_amdgcn_wmma_f32_16x16x32_f16(
            false, a, false, b, (short)0, cacc, false, false);
    }
    // combine softmax stats across the lane pair (l, l+16)
    float mo = __shfl_xor(m, 16);
    float so = __shfl_xor(s, 16);
    float M  = fmaxf(m, mo);
    float S  = s * __expf(m - M) + so * __expf(mo - M);
    float logZ = M + __logf(S);

    bool owner = (lane < 8) || (lane >= 24);
    if (owner) {
        int   idx = (lane < 8) ? lane : (lane - 24);
        float dot = cacc[idx];
        float ce  = Ssum[cls] * logZ - dot;   // = -sum_d W*(Xn - logZ)
        atomicAdd(&acc[cls], ce);
    }
}

// ---------------- kernel 6: per-class mean ------------------------------
__global__ void k_final(const float* __restrict__ acc, const int* __restrict__ counts,
                        float* __restrict__ out) {
    int i = blockIdx.x * blockDim.x + threadIdx.x;
    out[i] = acc[i] / (float)max(counts[i], 1);
}

// ---------------- host launcher -----------------------------------------
extern "C" void kernel_launch(void* const* d_in, const int* in_sizes, int n_in,
                              void* d_out, int out_size, void* d_ws, size_t ws_size,
                              hipStream_t stream) {
    const float* X   = (const float*)d_in[0];
    const int*   lbl = (const int*)d_in[1];
    int N = in_sizes[1];                     // 131072

    char*  ws      = (char*)d_ws;
    float* rnorm   = (float*)ws;                         // N floats
    int*   counts  = (int*)(ws + (size_t)N * 4);         // C ints
    int*   offsets = counts + CCLS;                      // C ints
    int*   fill    = offsets + CCLS;                     // C ints
    float* Ssum    = (float*)(fill + CCLS);              // C floats
    float* acc     = Ssum + CCLS;                        // C floats
    int*   rowidx  = (int*)(acc + CCLS);                 // N ints
    _Float16* Wh   = (_Float16*)(rowidx + N);            // C*D halves (32B aligned)
    float* out     = (float*)d_out;

    hipLaunchKernelGGL(k_init,       dim3(1),          dim3(CCLS), 0, stream, counts, acc);
    hipLaunchKernelGGL(k_norm_count, dim3(N / 8),      dim3(256),  0, stream, X, lbl, rnorm, counts);
    hipLaunchKernelGGL(k_scan,       dim3(1),          dim3(CCLS), 0, stream, counts, offsets, fill);
    hipLaunchKernelGGL(k_scatter,    dim3(N / 256),    dim3(256),  0, stream, lbl, fill, rowidx, N);
    hipLaunchKernelGGL(k_protos,     dim3(CCLS),       dim3(256),  0, stream, X, rnorm, rowidx, offsets, counts, Wh, Ssum);
    hipLaunchKernelGGL(k_ce,         dim3(N / 128),    dim3(256),  0, stream, X, lbl, rnorm, Wh, Ssum, acc);
    hipLaunchKernelGGL(k_final,      dim3(CCLS / 256), dim3(256),  0, stream, acc, counts, out);
}
